// GNNModel_40372692582493
// MI455X (gfx1250) — matrix operand order repaired and checked by reading, not compiled, not run on hardware
//
#include <hip/hip_runtime.h>
#include <hip/hip_bf16.h>
#include <math.h>

typedef __attribute__((ext_vector_type(16))) _Float16 v16h;
typedef __attribute__((ext_vector_type(8)))  _Float16 v8h;
typedef __attribute__((ext_vector_type(8)))  float    v8f;

#define LDT 40   // padded LDS row stride (halfs): 80B -> 16-way bank spread, 16B-aligned subrows

static inline unsigned gdivu(size_t a, int b) { return (unsigned)((a + (size_t)b - 1) / (size_t)b); }

// =============================================================
// Tiled WMMA GEMM: C[M,N] = act(A[M,K] @ B[K,N] + bias)
// fp32 in/out, f16 multiply via v_wmma_f32_16x16x32_f16.
// Block = 256 thr (8 waves), tile 64x64, wave grid 4x2; each wave
// computes a 16x32 strip = 2 WMMAs sharing one A fragment.
// Interior blocks stage with unguarded loads + single ds_store_b128
// per thread per operand; edges take the guarded path.
// ACT: 0=none, 1=relu
// =============================================================
template<int ACT>
__global__ __launch_bounds__(256)
void gemm_wmma_kernel(const float* __restrict__ A, const float* __restrict__ B,
                      const float* __restrict__ bias, float* __restrict__ C,
                      int M, int K, int N)
{
    __shared__ __align__(16) _Float16 Asm[64 * LDT];   // A tile 64x32 (row-major, padded)
    __shared__ __align__(16) _Float16 Bsm[64 * LDT];   // B tile transposed: Bsm[n][k]

    const int tid  = threadIdx.x;
    const int lane = tid & 31;
    const int wave = tid >> 5;
    const int wm   = wave >> 1;        // 0..3 -> row tile
    const int wn   = wave & 1;         // 0..1 -> 32-col strip
    const int row0 = blockIdx.y * 64;
    const int col0 = blockIdx.x * 64;
    const bool interior = (row0 + 64 <= M) && (col0 + 64 <= N);

    // staging mappings: one thread owns 8 contiguous K of one row/column
    const int ar  = tid >> 2;             // A row 0..63
    const int ak0 = (tid & 3) * 8;        // A k offset 0/8/16/24
    const int bn  = tid & 63;             // B col 0..63
    const int bk0 = (tid >> 6) * 8;       // B k offset 0/8/16/24

    const int mfrag = lane & 15;
    const int kA = (lane < 16) ? 0 : 16;   // A-frag K base, contiguous k = 16h + s (half-wave split)
    const int kB = (lane < 16) ? 0 : 16;   // B-frag K base, same contiguous order as A

    v8f acc0 = {}, acc1 = {};

    for (int k0 = 0; k0 < K; k0 += 32) {
        const bool fastk = interior && (k0 + 32 <= K);
        __syncthreads();

        // ---- stage A: 8 f32 loads -> v8h -> one ds_store_b128 ----
        {
            const int gr = row0 + ar;
            const float* ap = A + (size_t)gr * K + k0 + ak0;
            v8h av;
            if (fastk) {
                #pragma unroll
                for (int j = 0; j < 8; ++j) av[j] = (_Float16)ap[j];
            } else {
                #pragma unroll
                for (int j = 0; j < 8; ++j) {
                    int gk = k0 + ak0 + j;
                    av[j] = (gr < M && gk < K) ? (_Float16)ap[j] : (_Float16)0.0f;
                }
            }
            *(v8h*)&Asm[ar * LDT + ak0] = av;
        }
        // ---- stage B transposed: 8 stride-N loads (lane-coalesced) -> one ds_store_b128 ----
        {
            const int gc = col0 + bn;
            const float* bp = B + (size_t)(k0 + bk0) * N + gc;
            v8h bv;
            if (fastk) {
                #pragma unroll
                for (int j = 0; j < 8; ++j) bv[j] = (_Float16)bp[(size_t)j * N];
            } else {
                #pragma unroll
                for (int j = 0; j < 8; ++j) {
                    int gk = k0 + bk0 + j;
                    bv[j] = (gk < K && gc < N) ? (_Float16)bp[(size_t)j * N] : (_Float16)0.0f;
                }
            }
            *(v8h*)&Bsm[bn * LDT + bk0] = bv;
        }
        __syncthreads();

        // ---- fragments in contiguous k order, k = 16h + s (ds_load_b128 pairs) ----
        const int arow = wm * 16 + mfrag;
        v8h a0 = *(const v8h*)&Asm[arow * LDT + kA];
        v8h a1 = *(const v8h*)&Asm[arow * LDT + kA + 8];
        v16h af = __builtin_shufflevector(a0, a1, 0,1,2,3,4,5,6,7,8,9,10,11,12,13,14,15);

        const int bcol = wn * 32 + mfrag;
        v8h b00 = *(const v8h*)&Bsm[bcol * LDT + kB];
        v8h b01 = *(const v8h*)&Bsm[bcol * LDT + kB + 8];
        v16h bf0 = __builtin_shufflevector(b00, b01, 0,1,2,3,4,5,6,7,8,9,10,11,12,13,14,15);
        v8h b10 = *(const v8h*)&Bsm[(bcol + 16) * LDT + kB];
        v8h b11 = *(const v8h*)&Bsm[(bcol + 16) * LDT + kB + 8];
        v16h bf1 = __builtin_shufflevector(b10, b11, 0,1,2,3,4,5,6,7,8,9,10,11,12,13,14,15);

        acc0 = __builtin_amdgcn_wmma_f32_16x16x32_f16(false, af, false, bf0,
                                                      (short)0, acc0, false, false);
        acc1 = __builtin_amdgcn_wmma_f32_16x16x32_f16(false, af, false, bf1,
                                                      (short)0, acc1, false, false);
    }

    // ---- store 2 tiles ----
    #pragma unroll
    for (int tile = 0; tile < 2; ++tile) {
        const int col = col0 + wn * 32 + tile * 16 + (lane & 15);
        if (col < N) {
            float bv = bias ? bias[col] : 0.0f;
            const v8f& acc = tile ? acc1 : acc0;
            #pragma unroll
            for (int r = 0; r < 8; ++r) {
                int rowi = row0 + wm * 16 + r + ((lane < 16) ? 0 : 8);
                if (rowi < M) {
                    float v = acc[r] + bv;
                    if (ACT == 1) v = fmaxf(v, 0.0f);
                    C[(size_t)rowi * N + col] = v;
                }
            }
        }
    }
}

// =============================================================
// Utility fills (graph-capture safe; no hipMemset)
// =============================================================
__global__ void k_fill_f32(float* p, float v, long long n) {
    long long i = (long long)blockIdx.x * blockDim.x + threadIdx.x;
    if (i < n) p[i] = v;
}
__global__ void k_fill_u32(unsigned* p, unsigned v, long long n) {
    long long i = (long long)blockIdx.x * blockDim.x + threadIdx.x;
    if (i < n) p[i] = v;
}

// monotonic float<->uint encoding for atomicMax on floats
__device__ __forceinline__ unsigned encf(float f) {
    unsigned u = __float_as_uint(f);
    return (u & 0x80000000u) ? ~u : (u | 0x80000000u);
}
__device__ __forceinline__ float decf(unsigned u) {
    unsigned b = (u & 0x80000000u) ? (u ^ 0x80000000u) : ~u;
    return __uint_as_float(b);
}

// =============================================================
// GAT pieces
// =============================================================
__global__ void k_edge_count_sum(const int* __restrict__ src, const int* __restrict__ dst,
                                 const float* __restrict__ eattr,
                                 float* cnt, float* lsum, int E) {
    int e = blockIdx.x * blockDim.x + threadIdx.x;
    if (e >= E) return;
    int d = dst[e];
    (void)src;
    atomicAdd(&cnt[d], 1.0f);
    for (int f = 0; f < 27; ++f)
        atomicAdd(&lsum[(size_t)d * 27 + f], eattr[(size_t)e * 27 + f]);
}

__global__ void k_loop_attr_div(float* lsum, const float* cnt, int N) {
    int t = blockIdx.x * blockDim.x + threadIdx.x;
    if (t >= N * 27) return;
    int i = t / 27;
    lsum[t] /= fmaxf(cnt[i], 1.0f);
}

__global__ void k_attn_nodes(const float* __restrict__ h, const float* __restrict__ asrc,
                             const float* __restrict__ adst, float* a_s, float* a_d, int N) {
    int t = blockIdx.x * blockDim.x + threadIdx.x;
    if (t >= N * 4) return;
    int i = t >> 2, hh = t & 3;
    const float* hp = h + (size_t)i * 512 + hh * 128;
    const float* as = asrc + hh * 128;
    const float* ad = adst + hh * 128;
    float s1 = 0.f, s2 = 0.f;
    for (int d = 0; d < 128; ++d) { float hv = hp[d]; s1 += hv * as[d]; s2 += hv * ad[d]; }
    a_s[t] = s1; a_d[t] = s2;
}

// fold (ea @ We).reshape(E,H,D) . aedge  ->  ea @ M, M[f,h] = sum_d We[f,hD+d]*aedge[h,d]
__global__ void k_fold_We(const float* __restrict__ We, const float* __restrict__ aedge,
                          float* Mout) {
    int t = threadIdx.x;
    if (t >= 108) return;
    int f = t >> 2, hh = t & 3;
    float s = 0.f;
    for (int d = 0; d < 128; ++d) s += We[(size_t)f * 512 + hh * 128 + d] * aedge[hh * 128 + d];
    Mout[f * 4 + hh] = s;
}

__global__ void k_alpha(const int* __restrict__ src, const int* __restrict__ dst,
                        const float* __restrict__ eattr, const float* __restrict__ lattr,
                        const float* __restrict__ Mf, const float* __restrict__ a_s,
                        const float* __restrict__ a_d, float* alpha, int E, int N) {
    int e = blockIdx.x * blockDim.x + threadIdx.x;
    if (e >= E + N) return;
    int s, d; const float* ea;
    if (e < E) { s = src[e]; d = dst[e]; ea = eattr + (size_t)e * 27; }
    else       { s = d = e - E;          ea = lattr + (size_t)(e - E) * 27; }
    for (int hh = 0; hh < 4; ++hh) {
        float ae = 0.f;
        for (int f = 0; f < 27; ++f) ae += ea[f] * Mf[f * 4 + hh];
        float v = a_s[s * 4 + hh] + a_d[d * 4 + hh] + ae;
        alpha[(size_t)e * 4 + hh] = (v > 0.f) ? v : 0.2f * v;   // leaky_relu 0.2
    }
}

__global__ void k_segmax(const int* __restrict__ dst, const float* __restrict__ alpha,
                         unsigned* smax, int E, int N) {
    int e = blockIdx.x * blockDim.x + threadIdx.x;
    if (e >= E + N) return;
    int d = (e < E) ? dst[e] : (e - E);
    for (int hh = 0; hh < 4; ++hh)
        atomicMax(&smax[d * 4 + hh], encf(alpha[(size_t)e * 4 + hh]));
}

__global__ void k_segexp(const int* __restrict__ dst, float* alpha,
                         const unsigned* __restrict__ smax, float* ssum, int E, int N) {
    int e = blockIdx.x * blockDim.x + threadIdx.x;
    if (e >= E + N) return;
    int d = (e < E) ? dst[e] : (e - E);
    for (int hh = 0; hh < 4; ++hh) {
        float m = decf(smax[d * 4 + hh]);
        float w = expf(alpha[(size_t)e * 4 + hh] - m);
        alpha[(size_t)e * 4 + hh] = w;
        atomicAdd(&ssum[d * 4 + hh], w);
    }
}

__global__ void k_scatter_gat(const int* __restrict__ src, const int* __restrict__ dst,
                              const float* __restrict__ h, const float* __restrict__ alpha,
                              const float* __restrict__ ssum, float* acc, int E, int N) {
    int t = blockIdx.x * blockDim.x + threadIdx.x;
    int tot = (E + N) * 128;
    if (t >= tot) return;
    int e = t >> 7, d = t & 127;
    int s, dn;
    if (e < E) { s = src[e]; dn = dst[e]; } else { s = dn = e - E; }
    for (int hh = 0; hh < 4; ++hh) {
        float c = alpha[(size_t)e * 4 + hh] / (ssum[dn * 4 + hh] + 1e-16f);
        atomicAdd(&acc[(size_t)dn * 512 + hh * 128 + d],
                  h[(size_t)s * 512 + hh * 128 + d] * c);
    }
}

__global__ void k_gat_finish(const float* __restrict__ acc, const float* __restrict__ b,
                             float* xout, float* feat, int featOff, int N) {
    int t = blockIdx.x * blockDim.x + threadIdx.x;
    if (t >= N * 128) return;
    int i = t >> 7, d = t & 127;
    size_t base = (size_t)i * 512 + d;
    float v = 0.25f * (acc[base] + acc[base + 128] + acc[base + 256] + acc[base + 384]) + b[d];
    v = fmaxf(v, 0.f);
    xout[(size_t)i * 128 + d] = v;
    feat[(size_t)i * 551 + featOff + d] = v;
}

// =============================================================
// GCN pieces
// =============================================================
__global__ void k_dinv(const float* cnt, float* dinv, int N) {
    int i = blockIdx.x * blockDim.x + threadIdx.x;
    if (i < N) dinv[i] = rsqrtf(cnt[i] + 1.0f);   // deg includes self loop, always >=1
}

__global__ void k_scatter_gcn(const int* __restrict__ src, const int* __restrict__ dst,
                              const float* __restrict__ hg, const float* __restrict__ dinv,
                              float* acc, int E, int N) {
    int t = blockIdx.x * blockDim.x + threadIdx.x;
    int tot = (E + N) * 128;
    if (t >= tot) return;
    int e = t >> 7, d = t & 127;
    int s, dn;
    if (e < E) { s = src[e]; dn = dst[e]; } else { s = dn = e - E; }
    float nm = dinv[s] * dinv[dn];
    atomicAdd(&acc[(size_t)dn * 128 + d], hg[(size_t)s * 128 + d] * nm);
}

__global__ void k_gcn_finish(const float* __restrict__ acc, const float* __restrict__ b,
                             float* feat, int N) {
    int t = blockIdx.x * blockDim.x + threadIdx.x;
    if (t >= N * 128) return;
    int i = t >> 7, d = t & 127;
    feat[(size_t)i * 551 + 423 + d] = acc[(size_t)i * 128 + d] + b[d];
}

__global__ void k_copy_x(const float* __restrict__ x, float* feat, int N) {
    int t = blockIdx.x * blockDim.x + threadIdx.x;
    if (t >= N * 39) return;
    int i = t / 39, f = t % 39;
    feat[(size_t)i * 551 + f] = x[(size_t)i * 39 + f];
}

// =============================================================
// BatchNorm (training-mode batch stats, biased variance)
// =============================================================
__global__ __launch_bounds__(256)
void k_bn_stats(const float* __restrict__ h, int M, int Ncols, float* mean, float* rstd) {
    int col = blockIdx.x;
    float s = 0.f, s2 = 0.f;
    for (int r = threadIdx.x; r < M; r += 256) {
        float v = h[(size_t)r * Ncols + col];
        s += v; s2 += v * v;
    }
    __shared__ float sh[256], sh2[256];
    sh[threadIdx.x] = s; sh2[threadIdx.x] = s2;
    __syncthreads();
    for (int o = 128; o > 0; o >>= 1) {
        if ((int)threadIdx.x < o) { sh[threadIdx.x] += sh[threadIdx.x + o]; sh2[threadIdx.x] += sh2[threadIdx.x + o]; }
        __syncthreads();
    }
    if (threadIdx.x == 0) {
        float m = sh[0] / (float)M;
        float var = sh2[0] / (float)M - m * m;
        mean[col] = m;
        rstd[col] = rsqrtf(fmaxf(var, 0.f) + 1e-5f);
    }
}

__global__ void k_bn_apply(float* h, const float* __restrict__ mean, const float* __restrict__ rstd,
                           const float* __restrict__ g, const float* __restrict__ be,
                           long long total, int Ncols) {
    long long t = (long long)blockIdx.x * blockDim.x + threadIdx.x;
    if (t >= total) return;
    int col = (int)(t % Ncols);
    h[t] = (h[t] - mean[col]) * rstd[col] * g[col] + be[col];
}

// =============================================================
// Head feature builders
// =============================================================
__global__ void k_pairfeat(const int* __restrict__ i0, const int* __restrict__ i1,
                           const float* __restrict__ emb, const float* __restrict__ xyz,
                           const float* __restrict__ vec, float* Z, int Ep) {
    long long t = (long long)blockIdx.x * blockDim.x + threadIdx.x;
    long long tot = (long long)Ep * 514;
    if (t >= tot) return;
    int e = (int)(t / 514), j = (int)(t % 514);
    int u = i0[e], v = i1[e];
    float val;
    if (j < 256)      val = emb[(size_t)u * 256 + j];
    else if (j < 512) val = emb[(size_t)v * 256 + (j - 256)];
    else if (j == 512) {
        float dx = xyz[u * 3] - xyz[v * 3];
        float dy = xyz[u * 3 + 1] - xyz[v * 3 + 1];
        float dz = xyz[u * 3 + 2] - xyz[v * 3 + 2];
        val = dx * dx + dy * dy + dz * dz;            // squared distance (no sqrt)
    } else {
        float ax = vec[u * 3], ay = vec[u * 3 + 1], az = vec[u * 3 + 2];
        float bx = vec[v * 3], by = vec[v * 3 + 1], bz = vec[v * 3 + 2];
        float na = fmaxf(sqrtf(ax * ax + ay * ay + az * az), 1e-8f);
        float nb = fmaxf(sqrtf(bx * bx + by * by + bz * bz), 1e-8f);
        val = (ax * bx + ay * by + az * bz) / (na * nb);
    }
    Z[t] = val;
}

__global__ void k_catpair(const int* __restrict__ i0, const int* __restrict__ i1,
                          const float* __restrict__ emb, float* Z, int Ep) {
    long long t = (long long)blockIdx.x * blockDim.x + threadIdx.x;
    long long tot = (long long)Ep * 512;
    if (t >= tot) return;
    int e = (int)(t >> 9), j = (int)(t & 511);
    Z[t] = (j < 256) ? emb[(size_t)i0[e] * 256 + j] : emb[(size_t)i1[e] * 256 + (j - 256)];
}

// =============================================================
// Host-side helpers
// =============================================================
static void launch_gemm(const float* A, const float* B, const float* bias, float* C,
                        int M, int K, int N, int relu, hipStream_t stream) {
    dim3 blk(256);
    dim3 grd((N + 63) / 64, (M + 63) / 64);
    if (relu) gemm_wmma_kernel<1><<<grd, blk, 0, stream>>>(A, B, bias, C, M, K, N);
    else      gemm_wmma_kernel<0><<<grd, blk, 0, stream>>>(A, B, bias, C, M, K, N);
}

static void run_head(const float* Z, int Mh, int Kh,
                     const float* W1, const float* b1, const float* g, const float* be,
                     const float* W2, const float* b2, int Fo, float* out,
                     float* h1, float* bnm, float* bnr, hipStream_t stream) {
    launch_gemm(Z, W1, b1, h1, Mh, Kh, 1024, 1, stream);
    k_bn_stats<<<1024, 256, 0, stream>>>(h1, Mh, 1024, bnm, bnr);
    long long tot = (long long)Mh * 1024;
    k_bn_apply<<<gdivu((size_t)tot, 256), 256, 0, stream>>>(h1, bnm, bnr, g, be, tot, 1024);
    launch_gemm(h1, W2, b2, out, Mh, 1024, Fo, 0, stream);
}

extern "C" void kernel_launch(void* const* d_in, const int* in_sizes, int n_in,
                              void* d_out, int out_size, void* d_ws, size_t ws_size,
                              hipStream_t stream) {
    (void)n_in; (void)out_size; (void)ws_size;
    auto F = [&](int i) { return (const float*)d_in[i]; };
    auto I = [&](int i) { return (const int*)d_in[i]; };

    const float* x     = F(0);
    const int*   eidx  = I(1);
    const float* eattr = F(2);
    const int*   iep   = I(3);
    const int*   iei   = I(4);
    const float* xyz   = F(5);
    const float* vec   = F(6);
    const int*   a2bi  = I(7);

    const int N  = in_sizes[0] / 39;
    const int E  = in_sizes[1] / 2;
    const int EP = in_sizes[3] / 2;
    const int EI = in_sizes[4] / 2;
    const int EA = in_sizes[7] / 2;

    const int* src = eidx;
    const int* dst = eidx + E;

    // parameters: gat0 @9, gat1 @15, gat2 @21, gcn @27, fcn @29, link @35, a2b @41, node @47, int @53
    const float* gcnW = F(27);  const float* gcnb = F(28);

    // ---------------- workspace layout ----------------
    float* ws = (float*)d_ws;
    size_t off = 0;
    auto alloc = [&](size_t n) { float* p = ws + off; off += (n + 255) & ~(size_t)255; return p; };
    float* cnt   = alloc(N);
    float* dinv  = alloc(N);
    float* lattr = alloc((size_t)N * 27);
    float* h     = alloc((size_t)N * 512);
    float* xcur  = alloc((size_t)N * 128);
    float* a_s   = alloc((size_t)N * 4);
    float* a_d   = alloc((size_t)N * 4);
    float* Mfold = alloc(27 * 4);
    float* alpha = alloc((size_t)(E + N) * 4);
    float* smaxf = alloc((size_t)N * 4);   // used as unsigned
    float* ssum  = alloc((size_t)N * 4);
    float* acc   = alloc((size_t)N * 512); // reused by GCN (N*128)
    float* feat  = alloc((size_t)N * 551);
    float* hg    = alloc((size_t)N * 128);
    float* bnm   = alloc(1024);
    float* bnr   = alloc(1024);
    float* emb   = alloc((size_t)N * 256);
    float* h1    = alloc((size_t)EI * 1024);  // biggest head hidden
    float* Z     = alloc((size_t)EI * 514);   // biggest head input

    float* out = (float*)d_out;
    const size_t o_link = 0;
    const size_t o_a2b  = o_link + (size_t)EI * 1;
    const size_t o_node = o_a2b + (size_t)EA * 6;
    const size_t o_int  = o_node + (size_t)N * 16;

    // ---------------- self-loop attr (fill_value='mean') ----------------
    k_fill_f32<<<gdivu(N, 256), 256, 0, stream>>>(cnt, 0.f, N);
    k_fill_f32<<<gdivu((size_t)N * 27, 256), 256, 0, stream>>>(lattr, 0.f, (long long)N * 27);
    k_edge_count_sum<<<gdivu(E, 256), 256, 0, stream>>>(src, dst, eattr, cnt, lattr, E);
    k_loop_attr_div<<<gdivu((size_t)N * 27, 256), 256, 0, stream>>>(lattr, cnt, N);

    // ---------------- 3 GAT layers ----------------
    const float* xin = x;
    int fin = 39;
    for (int l = 0; l < 3; ++l) {
        int pb = 9 + l * 6;
        const float* W = F(pb), *We = F(pb + 1), *asr = F(pb + 2),
                   *ads = F(pb + 3), *aed = F(pb + 4), *bb = F(pb + 5);

        launch_gemm(xin, W, nullptr, h, N, fin, 512, 0, stream);
        k_attn_nodes<<<gdivu((size_t)N * 4, 256), 256, 0, stream>>>(h, asr, ads, a_s, a_d, N);
        k_fold_We<<<1, 128, 0, stream>>>(We, aed, Mfold);
        k_alpha<<<gdivu(E + N, 256), 256, 0, stream>>>(src, dst, eattr, lattr, Mfold,
                                                       a_s, a_d, alpha, E, N);
        // segment softmax over incoming edges
        k_fill_u32<<<gdivu((size_t)N * 4, 256), 256, 0, stream>>>((unsigned*)smaxf,
                                                                  0x007FFFFFu /* enc(-inf) */,
                                                                  (long long)N * 4);
        k_segmax<<<gdivu(E + N, 256), 256, 0, stream>>>(dst, alpha, (unsigned*)smaxf, E, N);
        k_fill_f32<<<gdivu((size_t)N * 4, 256), 256, 0, stream>>>(ssum, 0.f, (long long)N * 4);
        k_segexp<<<gdivu(E + N, 256), 256, 0, stream>>>(dst, alpha, (const unsigned*)smaxf,
                                                        ssum, E, N);
        k_fill_f32<<<gdivu((size_t)N * 512, 256), 256, 0, stream>>>(acc, 0.f, (long long)N * 512);
        k_scatter_gat<<<gdivu((size_t)(E + N) * 128, 256), 256, 0, stream>>>(src, dst, h, alpha,
                                                                             ssum, acc, E, N);
        k_gat_finish<<<gdivu((size_t)N * 128, 256), 256, 0, stream>>>(acc, bb, xcur, feat,
                                                                      39 + l * 128, N);
        xin = xcur; fin = 128;
    }
    k_copy_x<<<gdivu((size_t)N * 39, 256), 256, 0, stream>>>(x, feat, N);

    // ---------------- GCN ----------------
    k_dinv<<<gdivu(N, 256), 256, 0, stream>>>(cnt, dinv, N);
    launch_gemm(xcur, gcnW, nullptr, hg, N, 128, 128, 0, stream);
    k_fill_f32<<<gdivu((size_t)N * 128, 256), 256, 0, stream>>>(acc, 0.f, (long long)N * 128);
    k_scatter_gcn<<<gdivu((size_t)(E + N) * 128, 256), 256, 0, stream>>>(src, dst, hg, dinv,
                                                                         acc, E, N);
    k_gcn_finish<<<gdivu((size_t)N * 128, 256), 256, 0, stream>>>(acc, gcnb, feat, N);

    // ---------------- FCN: feat[551] -> 1024 -> BN -> 256 ----------------
    launch_gemm(feat, F(29), F(30), h1, N, 551, 1024, 1, stream);
    k_bn_stats<<<1024, 256, 0, stream>>>(h1, N, 1024, bnm, bnr);
    k_bn_apply<<<gdivu((size_t)N * 1024, 256), 256, 0, stream>>>(h1, bnm, bnr, F(31), F(32),
                                                                 (long long)N * 1024, 1024);
    launch_gemm(h1, F(33), F(34), emb, N, 1024, 256, 1, stream);

    // ---------------- heads ----------------
    // link: pair_feat(iei) [EI,514]
    k_pairfeat<<<gdivu((size_t)EI * 514, 256), 256, 0, stream>>>(iei, iei + EI, emb, xyz, vec, Z, EI);
    run_head(Z, EI, 514, F(35), F(36), F(37), F(38), F(39), F(40), 1, out + o_link,
             h1, bnm, bnr, stream);
    // a2b: concat(emb[i0],emb[i1]) [EA,512]
    k_catpair<<<gdivu((size_t)EA * 512, 256), 256, 0, stream>>>(a2bi, a2bi + EA, emb, Z, EA);
    run_head(Z, EA, 512, F(41), F(42), F(43), F(44), F(45), F(46), 6, out + o_a2b,
             h1, bnm, bnr, stream);
    // node: emb [N,256]
    run_head(emb, N, 256, F(47), F(48), F(49), F(50), F(51), F(52), 16, out + o_node,
             h1, bnm, bnr, stream);
    // int: pair_feat(iep) [EP,514]
    k_pairfeat<<<gdivu((size_t)EP * 514, 256), 256, 0, stream>>>(iep, iep + EP, emb, xyz, vec, Z, EP);
    run_head(Z, EP, 514, F(53), F(54), F(55), F(56), F(57), F(58), 3, out + o_int,
             h1, bnm, bnr, stream);
}
